// SelfAttention_9878424781059
// MI455X (gfx1250) — matrix operand order repaired and checked by reading, not compiled, not run on hardware
//
#include <hip/hip_runtime.h>
#include <math.h>

typedef __bf16 bf16_t;
typedef __attribute__((ext_vector_type(16))) __bf16 v16bf;
typedef __attribute__((ext_vector_type(8)))  __bf16 v8bf;
typedef __attribute__((ext_vector_type(8)))  float  v8f;

#define D_MODEL 1024
#define D_QUERY 64
#define T_SEQ   4096
#define BATCH   4
#define SM_SCALE 0.125f   // 1/sqrt(64)

// -------- WMMA helpers (gfx1250: V_WMMA_F32_16X16X32_BF16) --------

__device__ __forceinline__ v8f wmma_bf16(v16bf a, v16bf b, v8f c) {
  // 8 args: (neg_a, A, neg_b, B, c_mod, C, reuse_a, reuse_b)
  return __builtin_amdgcn_wmma_f32_16x16x32_bf16(false, a, false, b, (short)0, c,
                                                 false, false);
}

// A fragment (16x32 MxK bf16) from row-major [16 x >=32] src, leading dim ld.
// ISA layout: lane l -> row = l&15, khalf=(l>>4)*8; elems 0..7 -> K=khalf+e,
// elems 8..15 -> K=16+khalf+(e-8). Both halves are 16B-contiguous loads.
__device__ __forceinline__ v16bf load_a_bf16(const bf16_t* p, int ld) {
  const int l = threadIdx.x & 31;
  const bf16_t* r = p + (size_t)(l & 15) * ld + ((l >> 4) * 8);
  v8bf lo = *reinterpret_cast<const v8bf*>(r);
  v8bf hi = *reinterpret_cast<const v8bf*>(r + 16);
  v16bf out;
#pragma unroll
  for (int i = 0; i < 8; ++i) { out[i] = lo[i]; out[8 + i] = hi[i]; }
  return out;
}

// Same A layout but fp32 source, converting to bf16 on the fly.
__device__ __forceinline__ v16bf load_a_f32(const float* p, int ld) {
  const int l = threadIdx.x & 31;
  const float* r = p + (size_t)(l & 15) * ld + ((l >> 4) * 8);
  v16bf out;
#pragma unroll
  for (int i = 0; i < 8; ++i) {
    out[i]     = (bf16_t)r[i];
    out[8 + i] = (bf16_t)r[16 + i];
  }
  return out;
}

// B fragment (32x16 KxN bf16) loaded from B^T stored row-major [N x >=32]:
// lane l -> col n = l&15, elems 0..7 -> K = (l>>4)*8 + e, elems 8..15 -> K = 16 + (l>>4)*8 + (e-8).
__device__ __forceinline__ v16bf load_b_bf16(const bf16_t* pT, int ld) {
  const int l = threadIdx.x & 31;
  const bf16_t* r = pT + (size_t)(l & 15) * ld + ((l >> 4) * 8);
  v8bf lo = *reinterpret_cast<const v8bf*>(r);
  v8bf hi = *reinterpret_cast<const v8bf*>(r + 16);
  v16bf out;
#pragma unroll
  for (int i = 0; i < 8; ++i) { out[i] = lo[i]; out[8 + i] = hi[i]; }
  return out;
}

// -------- Kernel 1: W [1024 x N] fp32 -> Wt [N x 1024] bf16 --------
__global__ void wt_kernel(const float* __restrict__ W, bf16_t* __restrict__ Wt, int N) {
  int idx = blockIdx.x * blockDim.x + threadIdx.x;
  if (idx >= N * D_MODEL) return;
  int n = idx / D_MODEL;
  int k = idx - n * D_MODEL;
  Wt[idx] = (bf16_t)W[(size_t)k * N + n];  // Wt[n][k] = W[k][n]
}

// -------- Kernel 2: projection  X[M x 1024] @ W[1024 x N] -> bf16 --------
// One wave computes a 16(m) x 64(n) tile, K-loop over 1024.
// TRANSPOSE=0: Out[m][n] (ld=Nout).  TRANSPOSE=1: Vt[b][n][t] layout, with the
// 8 C-fragment rows per lane packed into a single 16B store (consecutive t).
template <int TRANSPOSE>
__global__ void proj_kernel(const float* __restrict__ X, const bf16_t* __restrict__ Wt,
                            bf16_t* __restrict__ Out, int Nout) {
  const int lane = threadIdx.x & 31;
  const int lh = lane >> 4, ln = lane & 15;
  const int mbase = blockIdx.x * 16;
  const int nbase = blockIdx.y * 64;
  const v8f zero8 = {0.f, 0.f, 0.f, 0.f, 0.f, 0.f, 0.f, 0.f};
  v8f acc[4];
#pragma unroll
  for (int nt = 0; nt < 4; ++nt) acc[nt] = zero8;

  for (int k = 0; k < D_MODEL; k += 32) {
    v16bf a = load_a_f32(X + (size_t)mbase * D_MODEL + k, D_MODEL);
#pragma unroll
    for (int nt = 0; nt < 4; ++nt) {
      v16bf bfr = load_b_bf16(Wt + (size_t)(nbase + nt * 16) * D_MODEL + k, D_MODEL);
      acc[nt] = wmma_bf16(a, bfr, acc[nt]);
    }
  }

  if (TRANSPOSE) {
    // Vt[b][n][t]; rows i are consecutive t (16-row tile never crosses batch).
    const int bb = mbase >> 12;
    const int t0 = (mbase & 4095) + lh * 8;  // first of 8 consecutive t
#pragma unroll
    for (int nt = 0; nt < 4; ++nt) {
      int n = nbase + nt * 16 + ln;
      v8bf pk;
#pragma unroll
      for (int i = 0; i < 8; ++i) pk[i] = (bf16_t)acc[nt][i];
      *reinterpret_cast<v8bf*>(Out + ((size_t)bb * Nout + n) * T_SEQ + t0) = pk;
    }
  } else {
#pragma unroll
    for (int nt = 0; nt < 4; ++nt) {
#pragma unroll
      for (int i = 0; i < 8; ++i) {
        int m = mbase + lh * 8 + i;          // C layout: row = 8*(l>>4)+i
        int n = nbase + nt * 16 + ln;        //           col = l&15
        Out[(size_t)m * Nout + n] = (bf16_t)acc[nt][i];
      }
    }
  }
}

// -------- Kernel 3: per-row softmax stats (max, denom) over causal keys ------
// One wave handles 32 query rows. S = Q@K^T via WMMA; online max/sum with
// xor-shuffle reductions across the 16-lane halves (wave32).
__global__ void stats_kernel(const bf16_t* __restrict__ Qb, const bf16_t* __restrict__ Kb,
                             float* __restrict__ smax, float* __restrict__ ssum) {
  const int lane = threadIdx.x & 31;
  const int lh = lane >> 4, ln = lane & 15;
  const int mblock = blockIdx.x;
  const int b = blockIdx.y;
  const int mbase = mblock * 32;
  const bf16_t* Q = Qb + (size_t)b * T_SEQ * D_QUERY;
  const bf16_t* K = Kb + (size_t)b * T_SEQ * D_QUERY;
  const v8f zero8 = {0.f, 0.f, 0.f, 0.f, 0.f, 0.f, 0.f, 0.f};

  v16bf qa[2][2];
#pragma unroll
  for (int mt = 0; mt < 2; ++mt)
#pragma unroll
    for (int kf = 0; kf < 2; ++kf)
      qa[mt][kf] = load_a_bf16(Q + (size_t)(mbase + mt * 16) * D_QUERY + kf * 32, D_QUERY);

  float mx[2][8], sm[2][8];
#pragma unroll
  for (int mt = 0; mt < 2; ++mt)
#pragma unroll
    for (int i = 0; i < 8; ++i) { mx[mt][i] = -3.0e38f; sm[mt][i] = 0.0f; }

  for (int nb = 0; nb <= mblock; ++nb) {
    const int nbase = nb * 32;
    v16bf kf_[2][2];
#pragma unroll
    for (int nt = 0; nt < 2; ++nt)
#pragma unroll
      for (int kf = 0; kf < 2; ++kf)
        kf_[nt][kf] = load_b_bf16(K + (size_t)(nbase + nt * 16) * D_QUERY + kf * 32, D_QUERY);

#pragma unroll
    for (int mt = 0; mt < 2; ++mt) {
      v8f s0 = zero8, s1 = zero8;
      s0 = wmma_bf16(qa[mt][0], kf_[0][0], s0);
      s0 = wmma_bf16(qa[mt][1], kf_[0][1], s0);
      s1 = wmma_bf16(qa[mt][0], kf_[1][0], s1);
      s1 = wmma_bf16(qa[mt][1], kf_[1][1], s1);
#pragma unroll
      for (int i = 0; i < 8; ++i) {
        int m = mbase + mt * 16 + lh * 8 + i;
        float z0 = ((nbase + ln) <= m)      ? s0[i] * SM_SCALE : -INFINITY;
        float z1 = ((nbase + 16 + ln) <= m) ? s1[i] * SM_SCALE : -INFINITY;
        float bm = fmaxf(z0, z1);
        bm = fmaxf(bm, __shfl_xor(bm, 1, 32));
        bm = fmaxf(bm, __shfl_xor(bm, 2, 32));
        bm = fmaxf(bm, __shfl_xor(bm, 4, 32));
        bm = fmaxf(bm, __shfl_xor(bm, 8, 32));
        float nm = fmaxf(mx[mt][i], bm);
        float e = __expf(z0 - nm) + __expf(z1 - nm);
        e += __shfl_xor(e, 1, 32);
        e += __shfl_xor(e, 2, 32);
        e += __shfl_xor(e, 4, 32);
        e += __shfl_xor(e, 8, 32);
        sm[mt][i] = sm[mt][i] * __expf(mx[mt][i] - nm) + e;
        mx[mt][i] = nm;
      }
    }
  }
  if (ln == 0) {
#pragma unroll
    for (int mt = 0; mt < 2; ++mt)
#pragma unroll
      for (int i = 0; i < 8; ++i) {
        int m = mbase + mt * 16 + lh * 8 + i;
        smax[(size_t)b * T_SEQ + m] = mx[mt][i];
        ssum[(size_t)b * T_SEQ + m] = sm[mt][i];
      }
  }
}

// -------- Kernel 4: O = softmax(QK^T) @ V for one 64-wide d_model slice ------
// One wave: 32 query rows x 64 output cols. Recompute S via WMMA, p=exp(.),
// bounce P through LDS (C-layout -> A-layout), WMMA against Vt (bf16).
__global__ void attn_out_kernel(const bf16_t* __restrict__ Qb, const bf16_t* __restrict__ Kb,
                                const bf16_t* __restrict__ Vt,
                                const float* __restrict__ smax, const float* __restrict__ ssum,
                                float* __restrict__ Out) {
  __shared__ __align__(16) bf16_t ptile[32][40];  // 32 rows x 32 keys, padded
  const int lane = threadIdx.x & 31;
  const int lh = lane >> 4, ln = lane & 15;
  const int mblock = blockIdx.x;   // T/32
  const int cslice = blockIdx.y;   // d_model/64
  const int b = blockIdx.z;
  const int mbase = mblock * 32;
  const int cbase = cslice * 64;
  const bf16_t* Q = Qb + (size_t)b * T_SEQ * D_QUERY;
  const bf16_t* K = Kb + (size_t)b * T_SEQ * D_QUERY;
  const bf16_t* V = Vt + (size_t)b * D_MODEL * T_SEQ;
  const v8f zero8 = {0.f, 0.f, 0.f, 0.f, 0.f, 0.f, 0.f, 0.f};

  v16bf qa[2][2];
#pragma unroll
  for (int mt = 0; mt < 2; ++mt)
#pragma unroll
    for (int kf = 0; kf < 2; ++kf)
      qa[mt][kf] = load_a_bf16(Q + (size_t)(mbase + mt * 16) * D_QUERY + kf * 32, D_QUERY);

  float zr[2][8], li[2][8];
#pragma unroll
  for (int mt = 0; mt < 2; ++mt)
#pragma unroll
    for (int i = 0; i < 8; ++i) {
      int m = mbase + mt * 16 + lh * 8 + i;
      zr[mt][i] = smax[(size_t)b * T_SEQ + m];
      li[mt][i] = 1.0f / ssum[(size_t)b * T_SEQ + m];
    }

  v8f o[2][4];
#pragma unroll
  for (int mt = 0; mt < 2; ++mt)
#pragma unroll
    for (int ct = 0; ct < 4; ++ct) o[mt][ct] = zero8;

  for (int nb = 0; nb <= mblock; ++nb) {
    const int nbase = nb * 32;
    v16bf kf_[2][2];
#pragma unroll
    for (int nt = 0; nt < 2; ++nt)
#pragma unroll
      for (int kf = 0; kf < 2; ++kf)
        kf_[nt][kf] = load_b_bf16(K + (size_t)(nbase + nt * 16) * D_QUERY + kf * 32, D_QUERY);

#pragma unroll
    for (int mt = 0; mt < 2; ++mt) {
#pragma unroll
      for (int nt = 0; nt < 2; ++nt) {
        v8f s = zero8;
        s = wmma_bf16(qa[mt][0], kf_[nt][0], s);
        s = wmma_bf16(qa[mt][1], kf_[nt][1], s);
#pragma unroll
        for (int i = 0; i < 8; ++i) {
          int m = mbase + mt * 16 + lh * 8 + i;
          int n = nbase + nt * 16 + ln;
          float p = (n <= m) ? __expf(s[i] * SM_SCALE - zr[mt][i]) : 0.0f;
          ptile[mt * 16 + lh * 8 + i][nt * 16 + ln] = (bf16_t)p;
        }
      }
    }
    __syncthreads();  // single-wave WG: degenerates to S_NOP, acts as LDS fence

#pragma unroll
    for (int mt = 0; mt < 2; ++mt) {
      v16bf pa = load_a_bf16(&ptile[mt * 16][0], 40);
#pragma unroll
      for (int ct = 0; ct < 4; ++ct) {
        v16bf vb = load_b_bf16(V + (size_t)(cbase + ct * 16) * T_SEQ + nbase, T_SEQ);
        o[mt][ct] = wmma_bf16(pa, vb, o[mt][ct]);
      }
    }
    __syncthreads();  // WAR guard before next iteration overwrites ptile
  }

#pragma unroll
  for (int mt = 0; mt < 2; ++mt)
#pragma unroll
    for (int ct = 0; ct < 4; ++ct)
#pragma unroll
      for (int i = 0; i < 8; ++i) {
        int m = mbase + mt * 16 + lh * 8 + i;
        int c = cbase + ct * 16 + ln;
        Out[((size_t)b * T_SEQ + m) * D_MODEL + c] = o[mt][ct][i] * li[mt][i];
      }
}

// -------- host launcher --------
extern "C" void kernel_launch(void* const* d_in, const int* in_sizes, int n_in,
                              void* d_out, int out_size, void* d_ws, size_t ws_size,
                              hipStream_t stream) {
  (void)in_sizes; (void)n_in; (void)out_size; (void)ws_size;
  const float* X  = (const float*)d_in[0];
  const float* Wq = (const float*)d_in[1];
  const float* Wk = (const float*)d_in[2];
  const float* Wv = (const float*)d_in[3];
  float* Out = (float*)d_out;

  char* ws = (char*)d_ws;
  size_t off = 0;
  bf16_t* Qb  = (bf16_t*)(ws + off); off += (size_t)BATCH * T_SEQ * D_QUERY * 2;   // 2 MB
  bf16_t* Kb  = (bf16_t*)(ws + off); off += (size_t)BATCH * T_SEQ * D_QUERY * 2;   // 2 MB
  bf16_t* Vt  = (bf16_t*)(ws + off); off += (size_t)BATCH * D_MODEL * T_SEQ * 2;   // 32 MB
  bf16_t* Wtq = (bf16_t*)(ws + off); off += (size_t)D_QUERY * D_MODEL * 2;
  bf16_t* Wtk = (bf16_t*)(ws + off); off += (size_t)D_QUERY * D_MODEL * 2;
  bf16_t* Wtv = (bf16_t*)(ws + off); off += (size_t)D_MODEL * D_MODEL * 2;         // 2 MB
  float*  smaxp = (float*)(ws + off); off += (size_t)BATCH * T_SEQ * 4;
  float*  ssump = (float*)(ws + off); off += (size_t)BATCH * T_SEQ * 4;

  // 1) weight transposes (fp32 -> bf16)
  wt_kernel<<<dim3((D_QUERY * D_MODEL + 255) / 256), 256, 0, stream>>>(Wq, Wtq, D_QUERY);
  wt_kernel<<<dim3((D_QUERY * D_MODEL + 255) / 256), 256, 0, stream>>>(Wk, Wtk, D_QUERY);
  wt_kernel<<<dim3((D_MODEL * D_MODEL + 255) / 256), 256, 0, stream>>>(Wv, Wtv, D_MODEL);

  // 2) projections (WMMA): Q,K row-major bf16; V stored transposed per batch
  const int Mrows = BATCH * T_SEQ;
  proj_kernel<0><<<dim3(Mrows / 16, D_QUERY / 64), 32, 0, stream>>>(X, Wtq, Qb, D_QUERY);
  proj_kernel<0><<<dim3(Mrows / 16, D_QUERY / 64), 32, 0, stream>>>(X, Wtk, Kb, D_QUERY);
  proj_kernel<1><<<dim3(Mrows / 16, D_MODEL / 64), 32, 0, stream>>>(X, Wtv, Vt, D_MODEL);

  // 3) softmax statistics (WMMA QK^T, causal, online max/sum)
  stats_kernel<<<dim3(T_SEQ / 32, BATCH), 32, 0, stream>>>(Qb, Kb, smaxp, ssump);

  // 4) output (WMMA QK^T recompute + WMMA P@V per 64-wide slice)
  attn_out_kernel<<<dim3(T_SEQ / 32, D_MODEL / 64, BATCH), 32, 0, stream>>>(
      Qb, Kb, Vt, smaxp, ssump, Out);
}
